// DeepMOI_40209483825577
// MI455X (gfx1250) — compile-verified, matrix operand order checked by reading, not run on hardware
//
#include <hip/hip_runtime.h>
#include <hip/hip_bf16.h>

// ---------------- problem constants (match reference) ----------------
#define GN    8000        // genes / nodes
#define GP    32          // pathways
#define GE    128000      // global edges
#define GES   4096        // edges per pathway
#define GSE   (GP*GES)    // 131072 subgraph edges
#define GDIN  16
#define GD    128
#define GPN   (GP*GN)     // 256000
#define GK1   6400
#define GK2   5120
#define GK3   4096

#define KPADMAX (128 + 8) // padded K row stride (halfs): 272 B rows, 16B aligned
#define MT    64          // output rows per GEMM block (4 x 16-row WMMA tiles)

typedef __attribute__((ext_vector_type(16))) _Float16 v16h;
typedef __attribute__((ext_vector_type(8)))  float    v8f;

// ---------------- tiny utility kernels ----------------
__global__ void k_fillf(float* p, float v, long n) {
    long i = (long)blockIdx.x * blockDim.x + threadIdx.x;
    if (i < n) p[i] = v;
}
__global__ void k_filli(int* p, int v, long n) {
    long i = (long)blockIdx.x * blockDim.x + threadIdx.x;
    if (i < n) p[i] = v;
}
__global__ void k_tile(const float* __restrict__ src, float* __restrict__ dst, long n, long period) {
    long i = (long)blockIdx.x * blockDim.x + threadIdx.x;
    if (i < n) dst[i] = src[i % period];
}

// ---------------- edge scatter (segment sum) ----------------
// one thread per (edge, feature); ew==null -> weight 1; deg==null -> 'add' aggregation
__global__ void k_scatter(const float* __restrict__ x, const int* __restrict__ src,
                          const int* __restrict__ dst, const float* __restrict__ ew,
                          float* agg, float* deg, int ne, int d) {
    long t = (long)blockIdx.x * blockDim.x + threadIdx.x;
    if (t >= (long)ne * d) return;
    int e = (int)(t / d), f = (int)(t % d);
    float w = ew ? ew[e] : 1.0f;
    if (w == 0.0f) return;
    int s = src[e], dd = dst[e];
    atomicAdd(&agg[(long)dd * d + f], x[(long)s * d + f] * w);
    if (f == 0 && deg) atomicAdd(&deg[dd], w);
}
// self loops (stage 1): unique (n,f) per thread, no atomics needed
__global__ void k_self(const float* __restrict__ x, float* agg, float* deg, long n, int d) {
    long t = (long)blockIdx.x * blockDim.x + threadIdx.x;
    if (t >= n * d) return;
    agg[t] += x[t];
    if ((t % d) == 0) deg[t / d] += 1.0f;
}
__global__ void k_div(float* agg, const float* __restrict__ deg, long n, int d) {
    long t = (long)blockIdx.x * blockDim.x + threadIdx.x;
    if (t >= n * d) return;
    agg[t] /= fmaxf(deg[t / d], 1.0f);
}

// ---------------- WMMA fused SAGE GEMM ----------------
// Y[M x 128] = act( A1[M x KD1] @ W1[KD1 x 128] + A2[M x KD2] @ W2[KD2 x 128] + bias )
// 1 block = 64 output rows (4 x 16-row tiles); 8 waves, wave w owns cols [16w,16w+16).
// A panel row-major [64 x KPAD] f16; W TRANSPOSED [128 cols x KPAD] f16 in LDS, so a
// lane's fragment elements are two contiguous 16B chunks -> ds_load_b128 for A and B.
// Each B fragment is reused across the 4 row tiles (4 WMMAs per fragment build).
// Zero-padding (K=16 -> 32) is a separate uniform pass: no divergent predication.
template <int KD>
__device__ __forceinline__ void gemm_op(const float* __restrict__ A,
                                        const float* __restrict__ W,
                                        long rowStart, int tid, int sub, int hi,
                                        int colBase, _Float16* sA, _Float16* sWt,
                                        v8f acc[4]) {
    constexpr int KP = (KD + 31) & ~31;   // zero-padded K extent
    if constexpr (KP != KD) {             // uniform zero-fill (K=16 case only)
        for (int t = tid; t < MT * KPADMAX; t += 256) sA[t] = (_Float16)0.0f;
        for (int t = tid; t < GD * KPADMAX; t += 256) sWt[t] = (_Float16)0.0f;
        __syncthreads();
    }
    // stage A panel [MT x KD] -> sA[r*KPAD + k], coalesced global reads
    for (int t = tid; t < MT * KD; t += 256) {
        int r = t / KD, k = t % KD;
        sA[r * KPADMAX + k] = (_Float16)A[rowStart * KD + t];
    }
    // stage W [KD x 128] -> transposed sWt[c*KPAD + k], coalesced global reads
    for (int t = tid; t < KD * GD; t += 256) {
        int k = t >> 7, c = t & (GD - 1);
        sWt[c * KPADMAX + k] = (_Float16)W[t];
    }
    __syncthreads();

    const int brow = (colBase + sub) * KPADMAX;
#pragma unroll
    for (int k0 = 0; k0 < KP; k0 += 32) {
        v16h bf;
#pragma unroll
        for (int i = 0; i < 8; ++i) {
            // 16x16x32 f16 fragment K-map (ISA 7.12.2): two 8-half contiguous chunks
            int kb = k0 + ((i < 4) ? (hi * 8 + 2 * i) : (16 + hi * 8 + 2 * (i - 4)));
            bf[2 * i]     = sWt[brow + kb];
            bf[2 * i + 1] = sWt[brow + kb + 1];
        }
#pragma unroll
        for (int rt = 0; rt < 4; ++rt) {
            const int arow = (rt * 16 + sub) * KPADMAX;
            v16h af;
#pragma unroll
            for (int i = 0; i < 8; ++i) {
                int kb = k0 + ((i < 4) ? (hi * 8 + 2 * i) : (16 + hi * 8 + 2 * (i - 4)));
                af[2 * i]     = sA[arow + kb];
                af[2 * i + 1] = sA[arow + kb + 1];
            }
            acc[rt] = __builtin_amdgcn_wmma_f32_16x16x32_f16(
                false, af, false, bf, (short)0, acc[rt], false, false);
        }
    }
    __syncthreads();   // LDS reused by next operand
}

template <int KD1, int KD2>
__global__ __launch_bounds__(256) void k_gemm2(
    const float* A1, const float* __restrict__ W1,
    const float* A2, const float* __restrict__ W2,
    const float* __restrict__ bias, float* Y, int act) {
    __shared__ _Float16 sWt[GD * KPADMAX];  // ~34 KB
    __shared__ _Float16 sA[MT * KPADMAX];   // ~17 KB
    const int tid  = threadIdx.x;
    const int lane = tid & 31;
    const int sub  = lane & 15;
    const int hi   = (lane >> 4) & 1;
    const int colBase   = (tid >> 5) * 16;
    const long rowStart = (long)blockIdx.x * MT;

    __builtin_prefetch(&W2[(long)(tid >> 1) * GD + (tid & 1) * 64], 0, 2); // warm W2 stream

    v8f acc[4] = {};
    gemm_op<KD1>(A1, W1, rowStart, tid, sub, hi, colBase, sA, sWt, acc);
    gemm_op<KD2>(A2, W2, rowStart, tid, sub, hi, colBase, sA, sWt, acc);

    const int c = colBase + sub;
    const float bc = bias[c];
#pragma unroll
    for (int rt = 0; rt < 4; ++rt) {
#pragma unroll
        for (int i = 0; i < 8; ++i) {
            long r  = rowStart + rt * 16 + i + hi * 8;  // C/D layout: VGPR i -> M=i / M=8+i
            float v = acc[rt][i] + bc;
            if (act) v = tanhf(v);
            Y[r * GD + c] = v;
        }
    }
}

// ---------------- GraphNorm (one block per feature column) ----------------
__global__ __launch_bounds__(256) void k_gnorm(float* x, const float* __restrict__ w,
                                               const float* __restrict__ b,
                                               const float* __restrict__ ms, int n) {
    __shared__ float red[256];
    __shared__ float s_mu, s_var;
    const int d = blockIdx.x, tid = threadIdx.x;
    float s = 0.0f;
    for (int r = tid; r < n; r += 256) s += x[(long)r * GD + d];
    red[tid] = s; __syncthreads();
    for (int o = 128; o > 0; o >>= 1) { if (tid < o) red[tid] += red[tid + o]; __syncthreads(); }
    if (!tid) s_mu = red[0] / n;
    __syncthreads();
    const float mu_ms = s_mu * ms[d];
    float v = 0.0f;
    for (int r = tid; r < n; r += 256) { float o = x[(long)r * GD + d] - mu_ms; v += o * o; }
    red[tid] = v; __syncthreads();
    for (int o = 128; o > 0; o >>= 1) { if (tid < o) red[tid] += red[tid + o]; __syncthreads(); }
    if (!tid) s_var = red[0] / n;
    __syncthreads();
    const float scale = w[d] * rsqrtf(s_var + 1e-5f);
    const float bb = b[d];
    for (int r = tid; r < n; r += 256) {
        long i = (long)r * GD + d;
        x[i] = scale * (x[i] - mu_ms) + bb;
    }
}

// ---------------- dual GEMV: out[i] = b + x1[i,:].w1 (+ x2[i,:].w2) ----------------
__global__ void k_gemv(const float* __restrict__ x1, const float* __restrict__ w1,
                       const float* __restrict__ x2, const float* __restrict__ w2,
                       const float* __restrict__ bptr, int bidx, float* out, long n) {
    long i = (long)blockIdx.x * blockDim.x + threadIdx.x;
    if (i >= n) return;
    float s = bptr ? bptr[bidx] : 0.0f;
    const float* r1 = x1 + i * GD;
#pragma unroll 8
    for (int f = 0; f < GD; ++f) s += r1[f] * w1[f];
    if (x2) {
        const float* r2 = x2 + i * GD;
#pragma unroll 8
        for (int f = 0; f < GD; ++f) s += r2[f] * w2[f];
    }
    out[i] = s;
}

// ---------------- global-attention readout (one block per graph) ----------------
__global__ __launch_bounds__(256) void k_readout(const float* __restrict__ x,
                                                 const float* __restrict__ gs,
                                                 const int* __restrict__ na,
                                                 float* out, int out_stride, int rows) {
    __shared__ float red[256];
    __shared__ float s_m, s_z;
    __shared__ float cols[2][GD];
    const int g = blockIdx.x, tid = threadIdx.x;
    const long base = (long)g * rows;
    float m = -3.4e38f;
    for (int r = tid; r < rows; r += 256) { long n = base + r; if (!na || na[n]) m = fmaxf(m, gs[n]); }
    red[tid] = m; __syncthreads();
    for (int o = 128; o > 0; o >>= 1) { if (tid < o) red[tid] = fmaxf(red[tid], red[tid + o]); __syncthreads(); }
    if (!tid) s_m = red[0];
    __syncthreads();
    float z = 0.0f;
    for (int r = tid; r < rows; r += 256) { long n = base + r; if (!na || na[n]) z += __expf(gs[n] - s_m); }
    red[tid] = z; __syncthreads();
    for (int o = 128; o > 0; o >>= 1) { if (tid < o) red[tid] += red[tid + o]; __syncthreads(); }
    if (!tid) s_z = red[0];
    __syncthreads();
    const int d = tid & (GD - 1), half = tid >> 7;
    float acc = 0.0f;
    for (int r = half; r < rows; r += 2) {
        long n = base + r;
        if (!na || na[n]) acc += __expf(gs[n] - s_m) * x[n * GD + d];
    }
    cols[half][d] = acc; __syncthreads();
    if (half == 0) out[(long)g * out_stride + d] = (cols[0][d] + cols[1][d]) / s_z;
}

// ---------------- SAGPool top-k: radix select per graph ----------------
__device__ __forceinline__ unsigned fkey(float f) {
    unsigned u = __float_as_uint(f);
    return (u & 0x80000000u) ? ~u : (u | 0x80000000u);
}
__global__ __launch_bounds__(256) void k_topk(const float* __restrict__ score,
                                              const int* __restrict__ na,
                                              unsigned* thr, int k, int rows) {
    __shared__ unsigned scnt;
    const int g = blockIdx.x, tid = threadIdx.x;
    const long base = (long)g * rows;
    unsigned prefix = 0; int kk = k;
    for (int bit = 31; bit >= 0; --bit) {
        const unsigned bm = 1u << bit;
        const unsigned ma = (bit == 31) ? 0u : (0xFFFFFFFFu << (bit + 1));
        if (!tid) scnt = 0;
        __syncthreads();
        unsigned c = 0;
        for (int r = tid; r < rows; r += 256) {
            long n = base + r;
            unsigned key = na[n] ? fkey(score[n]) : 0u;
            if ((key & ma) == prefix && (key & bm)) ++c;
        }
        atomicAdd(&scnt, c);
        __syncthreads();
        unsigned tot = scnt;
        __syncthreads();
        if ((int)tot >= kk) prefix |= bm; else kk -= (int)tot;
    }
    if (!tid) thr[g] = prefix;
}
__global__ void k_mark(const float* __restrict__ score, const int* __restrict__ na_old,
                       const unsigned* __restrict__ thr, int rows, int* na_new, long n) {
    long i = (long)blockIdx.x * blockDim.x + threadIdx.x;
    if (i >= n) return;
    int keep = na_old[i] && (fkey(score[i]) >= thr[i / rows]);
    na_new[i] = keep;
}
__global__ void k_apply(float* x, const float* __restrict__ score,
                        const int* __restrict__ na, long n128) {
    long t = (long)blockIdx.x * blockDim.x + threadIdx.x;
    if (t >= n128) return;
    long i = t >> 7;
    x[t] = na[i] ? x[t] * tanhf(score[i]) : 0.0f;
}
__global__ void k_eupd(float* ealive, const int* __restrict__ src, const int* __restrict__ dst,
                       const int* __restrict__ na, long ne) {
    long e = (long)blockIdx.x * blockDim.x + threadIdx.x;
    if (e >= ne) return;
    ealive[e] = (ealive[e] != 0.0f && na[src[e]] && na[dst[e]]) ? 1.0f : 0.0f;
}

// ---------------- final head ----------------
__global__ void k_mlp(const float* __restrict__ ro,
                      const float* __restrict__ lw, const float* __restrict__ lb,
                      const float* __restrict__ m1w, const float* __restrict__ m1b,
                      const float* __restrict__ m2w, const float* __restrict__ m2b,
                      const float* __restrict__ m3w, const float* __restrict__ m3b,
                      float* out) {
    __shared__ float t[GP + 1], h1[48], h2[16];
    const int tid = threadIdx.x;
    if (tid < GP + 1) {
        float s = lb[0];
        const float* r = ro + (long)tid * (3 * GD);
        for (int j = 0; j < 3 * GD; ++j) s += r[j] * lw[j];
        t[tid] = tanhf(s);
    }
    __syncthreads();
    if (tid < 48) {
        float s = m1b[tid];
        for (int i = 0; i < GP + 1; ++i) s += t[i] * m1w[i * 48 + tid];
        h1[tid] = tanhf(s);
    }
    __syncthreads();
    if (tid < 16) {
        float s = m2b[tid];
        for (int i = 0; i < 48; ++i) s += h1[i] * m2w[i * 16 + tid];
        h2[tid] = tanhf(s);
    }
    __syncthreads();
    if (!tid) {
        float s = m3b[0];
        for (int i = 0; i < 16; ++i) s += h2[i] * m3w[i];
        float a = 1.0f / (1.0f + __expf(-s));
        out[0] = 1.0f / (1.0f + __expf(-a));
    }
}

// ---------------- host orchestration ----------------
extern "C" void kernel_launch(void* const* d_in, const int* in_sizes, int n_in,
                              void* d_out, int out_size, void* d_ws, size_t ws_size,
                              hipStream_t stream) {
    (void)in_sizes; (void)n_in; (void)out_size; (void)ws_size;
    const float* h      = (const float*)d_in[0];
    const int*   ei     = (const int*)  d_in[1];   // [2,E] : src = ei, dst = ei+GE
    const int*   sei    = (const int*)  d_in[2];   // [2,SE]: ssrc = sei, sdst = sei+GSE
    const float* Wl_a   = (const float*)d_in[3];
    const float* Wr_a   = (const float*)d_in[4];
    const float* bl_a   = (const float*)d_in[5];
    const float* Wl_s   = (const float*)d_in[6];   // [5,128,128]
    const float* Wr_s   = (const float*)d_in[7];
    const float* bl_s   = (const float*)d_in[8];   // [5,128]
    const float* gate_w = (const float*)d_in[9];   // [6,128]
    const float* gate_b = (const float*)d_in[10];  // [6]
    const float* p_wrel = (const float*)d_in[11];  // [3,128]
    const float* p_brel = (const float*)d_in[12];  // [3]
    const float* p_wrt  = (const float*)d_in[13];  // [3,128]
    const float* norm_w = (const float*)d_in[14];
    const float* norm_b = (const float*)d_in[15];
    const float* norm_ms= (const float*)d_in[16];
    const float* lin_w  = (const float*)d_in[17];
    const float* lin_b  = (const float*)d_in[18];
    const float* m1w    = (const float*)d_in[19];
    const float* m1b    = (const float*)d_in[20];
    const float* m2w    = (const float*)d_in[21];
    const float* m2b    = (const float*)d_in[22];
    const float* m3w    = (const float*)d_in[23];
    const float* m3b    = (const float*)d_in[24];

    // carve workspace
    char* wsp = (char*)d_ws;
    auto carve = [&](size_t bytes) {
        void* p = (void*)wsp;
        wsp += (bytes + 255) & ~(size_t)255;
        return p;
    };
    float*    X      = (float*)   carve((size_t)GPN * GD * 4);
    float*    agg    = (float*)   carve((size_t)GPN * GD * 4);
    float*    x1     = (float*)   carve((size_t)GN * GD * 4);
    float*    deg    = (float*)   carve((size_t)GPN * 4);
    float*    score  = (float*)   carve((size_t)GPN * 4);
    float*    gs     = (float*)   carve((size_t)GPN * 4);
    int*      naA    = (int*)     carve((size_t)GPN * 4);
    int*      naB    = (int*)     carve((size_t)GPN * 4);
    float*    ealive = (float*)   carve((size_t)GSE * 4);
    unsigned* thr    = (unsigned*)carve((size_t)GP * 4);
    float*    ro     = (float*)   carve((size_t)(GP + 1) * 3 * GD * 4);

    const int T = 256;
    auto nb = [](long n, int t) { return (unsigned)((n + t - 1) / t); };
    const int DD = GD * GD;

    // ================= stage 1 : whole graph =================
    // layer a (DIN=16 input)
    k_fillf<<<nb((long)GN * GDIN, T), T, 0, stream>>>(agg, 0.f, (long)GN * GDIN);
    k_fillf<<<nb(GN, T), T, 0, stream>>>(deg, 0.f, GN);
    k_scatter<<<nb((long)GE * GDIN, T), T, 0, stream>>>(h, ei, ei + GE, nullptr, agg, deg, GE, GDIN);
    k_self<<<nb((long)GN * GDIN, T), T, 0, stream>>>(h, agg, deg, GN, GDIN);
    k_div<<<nb((long)GN * GDIN, T), T, 0, stream>>>(agg, deg, GN, GDIN);
    k_gemm2<GDIN, GDIN><<<GN / MT, T, 0, stream>>>(agg, Wl_a, h, Wr_a, bl_a, x1, 1);
    k_gemv<<<nb(GN, T), T, 0, stream>>>(x1, gate_w + 0 * GD, nullptr, nullptr, gate_b, 0, gs, GN);
    k_readout<<<1, T, 0, stream>>>(x1, gs, nullptr, ro + 0 * GD, 3 * GD, GN);
    k_gnorm<<<GD, T, 0, stream>>>(x1, norm_w, norm_b, norm_ms, GN);

    // layers b and c (stack indices 0 and 1; gate 1 and 2; readout parts 1 and 2)
    for (int L = 0; L < 2; ++L) {
        k_fillf<<<nb((long)GN * GD, T), T, 0, stream>>>(agg, 0.f, (long)GN * GD);
        k_fillf<<<nb(GN, T), T, 0, stream>>>(deg, 0.f, GN);
        k_scatter<<<nb((long)GE * GD, T), T, 0, stream>>>(x1, ei, ei + GE, nullptr, agg, deg, GE, GD);
        k_self<<<nb((long)GN * GD, T), T, 0, stream>>>(x1, agg, deg, GN, GD);
        k_div<<<nb((long)GN * GD, T), T, 0, stream>>>(agg, deg, GN, GD);
        k_gemm2<GD, GD><<<GN / MT, T, 0, stream>>>(agg, Wl_s + L * DD, x1, Wr_s + L * DD,
                                                   bl_s + L * GD, x1, 1);
        k_gemv<<<nb(GN, T), T, 0, stream>>>(x1, gate_w + (L + 1) * GD, nullptr, nullptr,
                                            gate_b, L + 1, gs, GN);
        k_readout<<<1, T, 0, stream>>>(x1, gs, nullptr, ro + (L + 1) * GD, 3 * GD, GN);
        if (L == 0)  // no GraphNorm after the last stage-1 layer
            k_gnorm<<<GD, T, 0, stream>>>(x1, norm_w, norm_b, norm_ms, GN);
    }

    // ================= stage 2 : P pathway subgraphs =================
    k_tile<<<nb((long)GPN * GD, T), T, 0, stream>>>(x1, X, (long)GPN * GD, (long)GN * GD);
    k_fillf<<<nb(GSE, T), T, 0, stream>>>(ealive, 1.f, GSE);
    k_filli<<<nb(GPN, T), T, 0, stream>>>(naA, 1, GPN);

    const int Ks[3] = {GK1, GK2, GK3};
    int* naCur  = naA;
    int* naNext = naB;
    for (int L = 0; L < 3; ++L) {
        const int si = 2 + L;          // stacked SAGE param index
        // ---- SAGE (mean over live edges) ----
        k_fillf<<<nb((long)GPN * GD, T), T, 0, stream>>>(agg, 0.f, (long)GPN * GD);
        k_fillf<<<nb(GPN, T), T, 0, stream>>>(deg, 0.f, GPN);
        k_scatter<<<nb((long)GSE * GD, T), T, 0, stream>>>(X, sei, sei + GSE, ealive, agg, deg, GSE, GD);
        k_div<<<nb((long)GPN * GD, T), T, 0, stream>>>(agg, deg, GPN, GD);
        k_gemm2<GD, GD><<<GPN / MT, T, 0, stream>>>(agg, Wl_s + si * DD, X, Wr_s + si * DD,
                                                    bl_s + si * GD, X, 1);
        // ---- SAGPool: GraphConv score ('add' aggregation), top-k, gate ----
        k_fillf<<<nb((long)GPN * GD, T), T, 0, stream>>>(agg, 0.f, (long)GPN * GD);
        k_scatter<<<nb((long)GSE * GD, T), T, 0, stream>>>(X, sei, sei + GSE, ealive, agg, nullptr, GSE, GD);
        k_gemv<<<nb(GPN, T), T, 0, stream>>>(agg, p_wrel + L * GD, X, p_wrt + L * GD,
                                             p_brel, L, score, GPN);
        k_topk<<<GP, T, 0, stream>>>(score, naCur, thr, Ks[L], GN);
        k_mark<<<nb(GPN, T), T, 0, stream>>>(score, naCur, thr, GN, naNext, GPN);
        k_apply<<<nb((long)GPN * GD, T), T, 0, stream>>>(X, score, naNext, (long)GPN * GD);
        k_eupd<<<nb(GSE, T), T, 0, stream>>>(ealive, sei, sei + GSE, naNext, GSE);
        // ---- masked global-attention readout (gate 3+L) ----
        k_gemv<<<nb(GPN, T), T, 0, stream>>>(X, gate_w + (3 + L) * GD, nullptr, nullptr,
                                             gate_b, 3 + L, gs, GPN);
        k_readout<<<GP, T, 0, stream>>>(X, gs, naNext, ro + 3 * GD /*skip row0*/ + L * GD,
                                        3 * GD, GN);
        // ping-pong alive masks
        int* tmp = naCur; naCur = naNext; naNext = tmp;
    }

    // ================= head =================
    k_mlp<<<1, 64, 0, stream>>>(ro, lin_w, lin_b, m1w, m1b, m2w, m2b, m3w, m3b, (float*)d_out);
}
